// SwitchTransformerMoE_16544214024869
// MI455X (gfx1250) — matrix operand order repaired and checked by reading, not compiled, not run on hardware
//
#include <hip/hip_runtime.h>
#include <hip/hip_bf16.h>
#include <stdint.h>

#define T_TOK  8192
#define DMODEL 1024
#define EXPERTS 8
#define HDIM   4096
#define CAPCT  1280   // int(1.25 * 8192 / 8)

typedef __attribute__((ext_vector_type(16))) __bf16 v16bf;
typedef __attribute__((ext_vector_type(8)))  float  v8f;
typedef unsigned short u16;
typedef unsigned int   u32;

union Frag { uint4 q[2]; v16bf v; };

__device__ __forceinline__ u16 f2bf(float f) {
  u32 u = __builtin_bit_cast(u32, f);
  u32 r = u + 0x7FFFu + ((u >> 16) & 1u);   // round-to-nearest-even
  return (u16)(r >> 16);
}
__device__ __forceinline__ u32 pk2bf(float lo, float hi) {
  return (u32)f2bf(lo) | ((u32)f2bf(hi) << 16);
}

// ---------------------------------------------------------------- init ----
__global__ __launch_bounds__(256) void init_kernel(float* __restrict__ out,
                                                   int* __restrict__ slot2tok) {
  int idx = blockIdx.x * 256 + threadIdx.x;              // 8192 blocks -> 2M idx
  float4 z; z.x = 0.f; z.y = 0.f; z.z = 0.f; z.w = 0.f;
  *(float4*)(out + (size_t)idx * 4) = z;                 // zero [T*D) output
  if (idx < EXPERTS * CAPCT) slot2tok[idx] = -1;
  if (idx < 9) out[(size_t)T_TOK * DMODEL + idx] = 0.f;  // entropy + counts
}

// -------------------------------------------------------------- router ----
__global__ __launch_bounds__(256) void router_kernel(
    const float* __restrict__ x, const float* __restrict__ rw,
    float* __restrict__ gate, int* __restrict__ eidx, float* __restrict__ ent,
    float* __restrict__ out_tail /* [0]=entropy (later), [1..8]=counts */) {
  int lane = threadIdx.x & 31;
  int w    = threadIdx.x >> 5;
  int t    = blockIdx.x * 8 + w;                 // one wave32 per token
  const float* xp = x + (size_t)t * DMODEL;

  float xs[32];
#pragma unroll
  for (int i = 0; i < 32; ++i) xs[i] = xp[i * 32 + lane];

  float logit[EXPERTS];
#pragma unroll
  for (int e = 0; e < EXPERTS; ++e) {
    const float* rp = rw + e * DMODEL;
    float s = 0.f;
#pragma unroll
    for (int i = 0; i < 32; ++i) s += xs[i] * rp[i * 32 + lane];
#pragma unroll
    for (int off = 16; off > 0; off >>= 1) s += __shfl_xor(s, off, 32);
    logit[e] = s;
  }
  float mx = logit[0];
#pragma unroll
  for (int e = 1; e < EXPERTS; ++e) mx = fmaxf(mx, logit[e]);
  float p[EXPERTS]; float se = 0.f;
#pragma unroll
  for (int e = 0; e < EXPERTS; ++e) { p[e] = __expf(logit[e] - mx); se += p[e]; }
  float inv = 1.f / se;
  float g = -1.f; int be = 0; float H = 0.f;
#pragma unroll
  for (int e = 0; e < EXPERTS; ++e) {
    p[e] *= inv;
    H -= p[e] * __logf(p[e] + 1e-8f);
    if (p[e] > g) { g = p[e]; be = e; }          // first-max wins, like argmax
  }
  if (lane == 0) {
    gate[t] = g; eidx[t] = be; ent[t] = H;
    atomicAdd(&out_tail[1 + be], 1.0f);          // integer-valued: exact & deterministic
  }
}

__global__ __launch_bounds__(256) void entropy_reduce_kernel(
    const float* __restrict__ ent, float* __restrict__ out_tail) {
  __shared__ float s[256];
  float a = 0.f;
  for (int i = threadIdx.x; i < T_TOK; i += 256) a += ent[i];  // fixed order
  s[threadIdx.x] = a; __syncthreads();
  for (int st = 128; st > 0; st >>= 1) {
    if (threadIdx.x < st) s[threadIdx.x] += s[threadIdx.x + st];
    __syncthreads();
  }
  if (threadIdx.x == 0) out_tail[0] = s[0] * (1.0f / T_TOK);
}

// ----------------------------------------------------- rank + dispatch ----
__global__ __launch_bounds__(256) void rank_dispatch_kernel(
    const float* __restrict__ x, const float* __restrict__ gate,
    const int* __restrict__ eidx, float* __restrict__ scale,
    int* __restrict__ slot2tok, u16* __restrict__ buf) {
  int t   = blockIdx.x;
  int tid = threadIdx.x;
  int   my_e = eidx[t];
  float my_g = gate[t];
  __shared__ int red[256];
  int cnt = 0;
  for (int j = tid; j < T_TOK; j += 256) {
    int   ej = eidx[j];
    float gj = gate[j];
    // stable lexsort (expert, -gate, index): rank within expert group
    if (ej == my_e && (gj > my_g || (gj == my_g && j < t))) ++cnt;
  }
  red[tid] = cnt; __syncthreads();
  for (int s = 128; s > 0; s >>= 1) {
    if (tid < s) red[tid] += red[tid + s];
    __syncthreads();
  }
  int  pos  = red[0];
  bool keep = pos < CAPCT;
  if (tid == 0) {
    scale[t] = keep ? my_g : 0.f;
    if (keep) slot2tok[my_e * CAPCT + pos] = t;
  }
  if (keep) {
    u16* dst = buf + ((size_t)my_e * CAPCT + pos) * DMODEL;
    const float* src = x + (size_t)t * DMODEL;
    int i = tid * 4;                              // 256 threads * 4 = D
    float4 v = *(const float4*)(src + i);
    uint2 pk;
    pk.x = pk2bf(v.x, v.y);
    pk.y = pk2bf(v.z, v.w);
    *(uint2*)(dst + i) = pk;
  }
}

// -------------------------------------------------- grouped WMMA GEMM ----
// C[e] = A[e] (bf16, MxK) * B[e] (fp32->bf16, KxN), software-pipelined
// global->reg->LDS staging, fused epilogue.
template <int KDIM, int NDIM, bool SECOND>
__global__ __launch_bounds__(256) void moe_gemm_kernel(
    const u16* __restrict__ A, const float* __restrict__ Bw,
    const float* __restrict__ bias,
    u16* __restrict__ Hout,                 // !SECOND: relu(acc+b1) -> bf16
    const int* __restrict__ slot2tok,       // SECOND: slot -> token
    const float* __restrict__ scale,        // SECOND: gate*keep per token
    float* __restrict__ out)                // SECOND: scatter to [T,D]
{
  constexpr int BM = 128, BN = 128, BK = 32;
  constexpr int LDA = 40;                   // bf16 stride: 80B, 16B-aligned, non-pow2
  __shared__ u16 sA[BM * LDA];
  __shared__ u16 sB[BN * LDA];              // transposed: sB[n*LDA + k]

  int e  = blockIdx.z;
  int m0 = blockIdx.y * BM;
  int n0 = blockIdx.x * BN;
  int tid  = threadIdx.x;
  int lane = tid & 31, w = tid >> 5;
  int wm = w & 1, wn = w >> 1;              // wave tile: 64(M) x 32(N)
  int half = lane >> 4, l16 = lane & 15;

  const u16*   Abase = A  + ((size_t)e * CAPCT + m0) * KDIM;
  const float* Bbase = Bw + (size_t)e * KDIM * NDIM + n0;

  // staging-thread mappings
  const int ar0 = (tid)       >> 2, ak0 = ((tid)       & 3) * 8;  // A piece 0
  const int ar1 = (tid + 256) >> 2, ak1 = ((tid + 256) & 3) * 8;  // A piece 1
  const int bn  = (tid & 63) * 2;                                 // B cols n, n+1
  const int bk  = (tid >> 6) * 8;                                 // B rows bk..bk+7

  uint4  aReg[2];
  float2 bReg[8];

  auto load_global = [&](int k0) {
    aReg[0] = *(const uint4*)(Abase + (size_t)ar0 * KDIM + (k0 + ak0));
    aReg[1] = *(const uint4*)(Abase + (size_t)ar1 * KDIM + (k0 + ak1));
#pragma unroll
    for (int i = 0; i < 8; ++i)
      bReg[i] = *(const float2*)(Bbase + (size_t)(k0 + bk + i) * NDIM + bn);
  };
  auto store_lds = [&]() {
    *(uint4*)(&sA[ar0 * LDA + ak0]) = aReg[0];
    *(uint4*)(&sA[ar1 * LDA + ak1]) = aReg[1];
    uint4 c0, c1;                               // pack K-pairs per column
    c0.x = pk2bf(bReg[0].x, bReg[1].x); c0.y = pk2bf(bReg[2].x, bReg[3].x);
    c0.z = pk2bf(bReg[4].x, bReg[5].x); c0.w = pk2bf(bReg[6].x, bReg[7].x);
    c1.x = pk2bf(bReg[0].y, bReg[1].y); c1.y = pk2bf(bReg[2].y, bReg[3].y);
    c1.z = pk2bf(bReg[4].y, bReg[5].y); c1.w = pk2bf(bReg[6].y, bReg[7].y);
    *(uint4*)(&sB[(bn + 0) * LDA + bk]) = c0;
    *(uint4*)(&sB[(bn + 1) * LDA + bk]) = c1;
  };

  v8f zero;
#pragma unroll
  for (int q = 0; q < 8; ++q) zero[q] = 0.f;
  v8f acc[4][2];
#pragma unroll
  for (int mi = 0; mi < 4; ++mi)
#pragma unroll
    for (int ni = 0; ni < 2; ++ni) acc[mi][ni] = zero;

  load_global(0);                               // prologue: stage tile 0

  for (int k0 = 0; k0 < KDIM; k0 += BK) {
    store_lds();
    __syncthreads();
    if (k0 + BK < KDIM) load_global(k0 + BK);   // prefetch next tile under compute

    // A frag: lanes 0-15 row=l16 hold K {ka..ka+7, ka+16..ka+23}, ka=half*8
    Frag a[4], b[2];
    int ka = half * 8;
#pragma unroll
    for (int mi = 0; mi < 4; ++mi) {
      int row = wm * 64 + mi * 16 + l16;
      a[mi].q[0] = *(const uint4*)(&sA[row * LDA + ka]);
      a[mi].q[1] = *(const uint4*)(&sA[row * LDA + ka + 16]);
    }
    // B frag: col=l16 holds K {kb..kb+7, kb+16..kb+23}, kb=half*8
    int kb = half * 8;
#pragma unroll
    for (int ni = 0; ni < 2; ++ni) {
      int col = wn * 32 + ni * 16 + l16;
      b[ni].q[0] = *(const uint4*)(&sB[col * LDA + kb]);
      b[ni].q[1] = *(const uint4*)(&sB[col * LDA + kb + 16]);
    }
#pragma unroll
    for (int mi = 0; mi < 4; ++mi)
#pragma unroll
      for (int ni = 0; ni < 2; ++ni)
        acc[mi][ni] = __builtin_amdgcn_wmma_f32_16x16x32_bf16(
            false, a[mi].v, false, b[ni].v, (short)0, acc[mi][ni], false, false);
    __syncthreads();
  }

  // epilogue: C layout — lane l16 = col, vgpr j -> row (half*8 + j)
#pragma unroll
  for (int mi = 0; mi < 4; ++mi) {
    int mbase = m0 + wm * 64 + mi * 16 + half * 8;
    if (!SECOND) {
#pragma unroll
      for (int ni = 0; ni < 2; ++ni) {
        int n = n0 + wn * 32 + ni * 16 + l16;
        float bv = bias[e * NDIM + n];
#pragma unroll
        for (int j = 0; j < 8; ++j) {
          float vv = fmaxf(acc[mi][ni][j] + bv, 0.f);   // ReLU
          Hout[((size_t)e * CAPCT + (mbase + j)) * NDIM + n] = f2bf(vv);
        }
      }
    } else {
      int   toks[8];
      float sc[8];
#pragma unroll
      for (int j = 0; j < 8; ++j) {
        toks[j] = slot2tok[e * CAPCT + mbase + j];
        sc[j]   = (toks[j] >= 0) ? scale[toks[j]] : 0.f;
      }
#pragma unroll
      for (int ni = 0; ni < 2; ++ni) {
        int n = n0 + wn * 32 + ni * 16 + l16;
        float bv = bias[e * NDIM + n];
#pragma unroll
        for (int j = 0; j < 8; ++j) {
          if (toks[j] >= 0)
            out[(size_t)toks[j] * NDIM + n] = (acc[mi][ni][j] + bv) * sc[j];
        }
      }
    }
  }
}

// --------------------------------------------------------------- launch ----
extern "C" void kernel_launch(void* const* d_in, const int* in_sizes, int n_in,
                              void* d_out, int out_size, void* d_ws, size_t ws_size,
                              hipStream_t stream) {
  const float* x        = (const float*)d_in[0];
  const float* router_w = (const float*)d_in[1];
  const float* W1       = (const float*)d_in[2];
  const float* b1       = (const float*)d_in[3];
  const float* W2       = (const float*)d_in[4];
  const float* b2       = (const float*)d_in[5];
  float* out = (float*)d_out;

  char* ws = (char*)d_ws;
  float* gate     = (float*)(ws + 0);          // T*4
  int*   eidx     = (int*)  (ws + 32768);      // T*4
  float* scale    = (float*)(ws + 65536);      // T*4
  float* ent      = (float*)(ws + 98304);      // T*4
  int*   slot2tok = (int*)  (ws + 131072);     // E*CAP*4 = 40960
  u16*   buf      = (u16*)  (ws + 172032);     // E*CAP*D*2 = 20,971,520
  u16*   hbuf     = (u16*)  (ws + 21143552);   // E*CAP*H*2 = 83,886,080

  float* out_tail = out + (size_t)T_TOK * DMODEL;

  init_kernel<<<8192, 256, 0, stream>>>(out, slot2tok);
  router_kernel<<<T_TOK / 8, 256, 0, stream>>>(x, router_w, gate, eidx, ent, out_tail);
  rank_dispatch_kernel<<<T_TOK, 256, 0, stream>>>(x, gate, eidx, scale, slot2tok, buf);
  entropy_reduce_kernel<<<1, 256, 0, stream>>>(ent, out_tail);

  moe_gemm_kernel<DMODEL, HDIM, false>
      <<<dim3(HDIM / 128, CAPCT / 128, EXPERTS), 256, 0, stream>>>(
          buf, W1, b1, hbuf, nullptr, nullptr, nullptr);
  moe_gemm_kernel<HDIM, DMODEL, true>
      <<<dim3(DMODEL / 128, CAPCT / 128, EXPERTS), 256, 0, stream>>>(
          hbuf, W2, b2, nullptr, slot2tok, scale, out);
}
